// BrainGNN_81784767250575
// MI455X (gfx1250) — compile-verified, matrix operand order checked by reading, not run on hardware
//
#include <hip/hip_runtime.h>
#include <hip/hip_bf16.h>
#include <math.h>

// ---------------- problem constants (static per reference) ----------------
#define BG   256          // graphs
#define RR   268          // ROIs
#define DEG  16
#define HH   128          // hidden
#define CC   7            // communities
#define N0   (BG * RR)    // 68608 nodes layer 0
#define EE   (N0 * DEG)   // 1097728 edges
#define KP0  288          // K=268 padded to multiple of 32 for the input GEMM

typedef __attribute__((ext_vector_type(16))) __bf16 v16bf;
typedef __attribute__((ext_vector_type(8)))  __bf16 v8bf;
typedef __attribute__((ext_vector_type(8)))  float  v8f;

union V16U { v16bf v; v8bf h[2]; };

// ---------------- small device helpers ----------------
static __device__ __forceinline__ __bf16 f2bf(float f) {
  unsigned u = __float_as_uint(f);
  unsigned r = u + 0x7FFFu + ((u >> 16) & 1u);   // round-to-nearest-even
  unsigned short h = (unsigned short)(r >> 16);
  return __builtin_bit_cast(__bf16, h);
}
static __device__ __forceinline__ float bf2f(__bf16 h) {
  unsigned short u = __builtin_bit_cast(unsigned short, h);
  return __uint_as_float(((unsigned)u) << 16);
}
static __device__ __forceinline__ float sigmoidf_(float x) { return 1.0f / (1.0f + expf(-x)); }
static __device__ __forceinline__ float eluf_(float x)     { return x > 0.0f ? x : expf(x) - 1.0f; }

// K index inside a 16x16x32 bf16 fragment (ISA 7.12.2):
// lane&15 = M(A)/N(B), koff=(lane>>4)*8; e<8 -> K=koff+e ; e>=8 -> K=8+koff+e
static __device__ __forceinline__ int frag_k(int koff, int e) {
  return (e < 8) ? (koff + e) : (8 + koff + e);
}

// =====================================================================
// Pack a row-major f32 weight (K x Nc) into bf16 WMMA B-fragment order:
// P[fi=nt*ksteps+ks][lane][e]  (512 bf16 per fragment, lane-contiguous)
// =====================================================================
__global__ void bgnn_pack_b(const float* __restrict__ W, __bf16* __restrict__ P,
                            int K, int Kp, int Nc) {
  const int ksteps = Kp >> 5;
  const int fi = blockIdx.x;
  const int nt = fi / ksteps, ks = fi % ksteps;
  const int lane = threadIdx.x;
  const int koff = (lane >> 4) * 8;
  const int col = nt * 16 + (lane & 15);
  __bf16* out = P + (size_t)fi * 512 + (size_t)lane * 16;
#pragma unroll
  for (int e = 0; e < 16; ++e) {
    int kk = ks * 32 + frag_k(koff, e);
    float v = (kk < K) ? W[(size_t)kk * Nc + col] : 0.0f;
    out[e] = f2bf(v);
  }
}

// ---------------- convert x (N0 x 268 f32) -> XA (N0 x 288 bf16, zero-padded) ----------------
__global__ void bgnn_cvt_x(const float* __restrict__ x, __bf16* __restrict__ XA) {
  int r = blockIdx.x, t = threadIdx.x;          // blockDim = KP0 = 288
  float v = (t < RR) ? x[(size_t)r * RR + t] : 0.0f;
  XA[(size_t)r * KP0 + t] = f2bf(v);
}

// =====================================================================
// Row-panel WMMA GEMM: C[M,Nc](f32) = act(Abf[M,Kp] @ Bpacked + bias[Nc])
// grid.x = M/16, block = 32 (one wave). Wave owns 16 rows x all Nc cols:
// A strip read once per wave (L0 resident on reload), B streamed from L2.
// ACT: 0=none, 1=tanh, 2=ELU
// =====================================================================
template <int ACT>
__global__ void bgnn_wmma_gemm(const __bf16* __restrict__ A, const __bf16* __restrict__ Bp,
                               const float* __restrict__ bias, float* __restrict__ Cout,
                               int Kp, int Nc) {
  const int mt = blockIdx.x;
  const int lane = threadIdx.x;
  const int koff = (lane >> 4) * 8;
  const int ksteps = Kp >> 5;
  const int ntiles = Nc >> 4;
  const __bf16* arow = A + (size_t)(mt * 16 + (lane & 15)) * Kp + koff;
  const __bf16* bl   = Bp + (size_t)lane * 16;
  const int rbase = mt * 16 + (lane >> 4) * 8;
#pragma unroll 1
  for (int nt = 0; nt < ntiles; ++nt) {
    v8f acc = {};
#pragma unroll 1
    for (int ks = 0; ks < ksteps; ++ks) {
      V16U a, b;
      a.h[0] = *(const v8bf*)(arow + ks * 32);        // global_load_b128
      a.h[1] = *(const v8bf*)(arow + ks * 32 + 16);   // global_load_b128
      b.v    = *(const v16bf*)(bl + (size_t)(nt * ksteps + ks) * 512);
      acc = __builtin_amdgcn_wmma_f32_16x16x32_bf16(false, a.v, false, b.v, (short)0, acc,
                                                    false, false);
    }
    const int ocol = nt * 16 + (lane & 15);
    float bs = bias[ocol];
#pragma unroll
    for (int i = 0; i < 8; ++i) {
      float v = acc[i] + bs;
      if (ACT == 1) v = tanhf(v);
      if (ACT == 2) v = eluf_(v);
      Cout[(size_t)(rbase + i) * Nc + ocol] = v;
    }
  }
}

// =====================================================================
// ROI-aware transform (row panel, K=N=128):
//   ht[n] = sum_c cw[n%R, c] * (Hin[n] @ basis[c])
// A strip kept in registers (4 fragments), reused across 8 n-tiles x 7 bases.
// Writes raw ht (f32, edge messages) and out = ht*sigmoid(We+be)+bias.
// =====================================================================
__global__ void bgnn_wmma_transform(const __bf16* __restrict__ Hin,
                                    const __bf16* __restrict__ Pbasis, // 7 packed 128x128
                                    const float* __restrict__ cw,
                                    const float* __restrict__ We, const float* __restrict__ be,
                                    const float* __restrict__ bias,
                                    float* __restrict__ Ht, float* __restrict__ Out,
                                    int writeHt) {
  const int mt = blockIdx.x;
  const int lane = threadIdx.x;
  const int koff = (lane >> 4) * 8;
  const int rbase = mt * 16 + (lane >> 4) * 8;
  const __bf16* arow = Hin + (size_t)(mt * 16 + (lane & 15)) * HH + koff;
  const __bf16* bl   = Pbasis + (size_t)lane * 16;

  V16U afrag[4];
#pragma unroll
  for (int ks = 0; ks < 4; ++ks) {
    afrag[ks].h[0] = *(const v8bf*)(arow + ks * 32);
    afrag[ks].h[1] = *(const v8bf*)(arow + ks * 32 + 16);
  }
  int nids[8];
#pragma unroll
  for (int i = 0; i < 8; ++i) nids[i] = (rbase + i) % RR;

#pragma unroll 1
  for (int nt = 0; nt < 8; ++nt) {
    float tot[8];
#pragma unroll
    for (int i = 0; i < 8; ++i) tot[i] = 0.0f;
#pragma unroll 1
    for (int c = 0; c < CC; ++c) {
      const __bf16* bc = bl + (size_t)(c * 32 + nt * 4) * 512;  // c*(ksteps*ntiles) + nt*ksteps
      v8f acc = {};
#pragma unroll
      for (int ks = 0; ks < 4; ++ks) {
        V16U b;
        b.v = *(const v16bf*)(bc + (size_t)ks * 512);
        acc = __builtin_amdgcn_wmma_f32_16x16x32_bf16(false, afrag[ks].v, false, b.v,
                                                      (short)0, acc, false, false);
      }
#pragma unroll
      for (int i = 0; i < 8; ++i) tot[i] += cw[nids[i] * CC + c] * acc[i];
    }
    const int ocol = nt * 16 + (lane & 15);
    float sw = sigmoidf_(We[ocol] + be[ocol]);   // self-loop edge weight at attr=1
    float bb = bias[ocol];
#pragma unroll
    for (int i = 0; i < 8; ++i) {
      int r = rbase + i;
      if (writeHt) Ht[(size_t)r * HH + ocol] = tot[i];
      Out[(size_t)r * HH + ocol] = tot[i] * sw + bb;
    }
  }
}

// =====================================================================
// Fused attention score: S[r] = tanh(A[r]@W1 + b1) . w2 + b2[0]
// Row-panel wave holds all 128 cols of its 16 rows -> per-lane partial dot,
// then __shfl_xor reduction across the 16-lane column group. K=N=128.
// =====================================================================
__global__ void bgnn_wmma_score(const __bf16* __restrict__ A, const __bf16* __restrict__ Bp,
                                const float* __restrict__ b1, const float* __restrict__ w2,
                                const float* __restrict__ b2, float* __restrict__ S) {
  const int mt = blockIdx.x;
  const int lane = threadIdx.x;
  const int koff = (lane >> 4) * 8;
  const int rbase = mt * 16 + (lane >> 4) * 8;
  const __bf16* arow = A + (size_t)(mt * 16 + (lane & 15)) * HH + koff;
  const __bf16* bl   = Bp + (size_t)lane * 16;

  V16U afrag[4];
#pragma unroll
  for (int ks = 0; ks < 4; ++ks) {
    afrag[ks].h[0] = *(const v8bf*)(arow + ks * 32);
    afrag[ks].h[1] = *(const v8bf*)(arow + ks * 32 + 16);
  }
  float dsum[8];
#pragma unroll
  for (int i = 0; i < 8; ++i) dsum[i] = 0.0f;

#pragma unroll 1
  for (int nt = 0; nt < 8; ++nt) {
    v8f acc = {};
#pragma unroll
    for (int ks = 0; ks < 4; ++ks) {
      V16U b;
      b.v = *(const v16bf*)(bl + (size_t)(nt * 4 + ks) * 512);
      acc = __builtin_amdgcn_wmma_f32_16x16x32_bf16(false, afrag[ks].v, false, b.v,
                                                    (short)0, acc, false, false);
    }
    const int ocol = nt * 16 + (lane & 15);
    float bs = b1[ocol], wv = w2[ocol];
#pragma unroll
    for (int i = 0; i < 8; ++i) dsum[i] += tanhf(acc[i] + bs) * wv;
  }
  // reduce across the 16 lanes holding the same rows (xor masks 1,2,4,8)
#pragma unroll
  for (int m = 8; m > 0; m >>= 1) {
#pragma unroll
    for (int i = 0; i < 8; ++i) dsum[i] += __shfl_xor(dsum[i], m);
  }
  if ((lane & 15) == 0) {
    float bb = b2[0];
#pragma unroll
    for (int i = 0; i < 8; ++i) S[rbase + i] = dsum[i] + bb;
  }
}

// ---------------- community softmax: cw[r,c] = softmax_c(roi[r,c]) ----------------
__global__ void bgnn_row_softmax7(const float* __restrict__ roi, float* __restrict__ cw) {
  int r = blockIdx.x * blockDim.x + threadIdx.x;
  if (r >= RR) return;
  float m = -INFINITY;
  for (int c = 0; c < CC; ++c) m = fmaxf(m, roi[r * CC + c]);
  float s = 0.0f, e[CC];
  for (int c = 0; c < CC; ++c) { e[c] = expf(roi[r * CC + c] - m); s += e[c]; }
  float inv = 1.0f / s;
  for (int c = 0; c < CC; ++c) cw[r * CC + c] = e[c] * inv;
}

// ---------------- edge aggregation: Out[dst] += ht[src] * sigmoid(attr*We+be) ----------------
__global__ void bgnn_edge_agg(const int* __restrict__ ei, const float* __restrict__ eattr,
                              const float* __restrict__ Ht,
                              const float* __restrict__ We, const float* __restrict__ be,
                              float* __restrict__ Out) {
  int e = blockIdx.x;
  int f = threadIdx.x;
  int src = ei[e];
  int dst = ei[EE + e];
  float w = sigmoidf_(eattr[e] * We[f] + be[f]);
  atomicAdd(&Out[(size_t)dst * HH + f], Ht[(size_t)src * HH + f] * w);
}

// ---------------- row LayerNorm+ELU, f32 in -> bf16 out ----------------
// MODE 0: elu(ln(x)) ; MODE 1: ln(elu(x))
template <int MODE>
__global__ void bgnn_actln(const float* __restrict__ X, const float* __restrict__ g,
                           const float* __restrict__ b, __bf16* __restrict__ Y) {
  __shared__ float red[HH];
  int r = blockIdx.x, t = threadIdx.x;
  float v = X[(size_t)r * HH + t];
  if (MODE == 1) v = eluf_(v);
  red[t] = v; __syncthreads();
  for (int s = HH / 2; s > 0; s >>= 1) { if (t < s) red[t] += red[t + s]; __syncthreads(); }
  float m = red[0] * (1.0f / HH); __syncthreads();
  float d = v - m;
  red[t] = d * d; __syncthreads();
  for (int s = HH / 2; s > 0; s >>= 1) { if (t < s) red[t] += red[t + s]; __syncthreads(); }
  float var = red[0] * (1.0f / HH); __syncthreads();
  float y = d * rsqrtf(var + 1e-5f) * g[t] + b[t];
  if (MODE == 0) y = eluf_(y);
  Y[(size_t)r * HH + t] = f2bf(y);
}

// ---------------- per-row dot: out[r] = X[r,:K].w + bscal[0] (blockDim == K) ----------------
__global__ void bgnn_rowdot(const float* __restrict__ X, const float* __restrict__ w,
                            const float* __restrict__ bscal, float* __restrict__ out, int K) {
  __shared__ float red[HH];
  int r = blockIdx.x, t = threadIdx.x;
  red[t] = X[(size_t)r * K + t] * w[t]; __syncthreads();
  for (int s = blockDim.x / 2; s > 0; s >>= 1) { if (t < s) red[t] += red[t + s]; __syncthreads(); }
  if (t == 0) out[r] = red[0] + bscal[0];
}

// ---------------- per-graph static top-k (desc, ties -> lowest index first) ----------------
__global__ void bgnn_topk(const float* __restrict__ S, int n, int k,
                          int* __restrict__ Idx, float* __restrict__ Sv) {
  __shared__ float sv[272];
  __shared__ float rv[128];
  __shared__ int   ri[128];
  int b = blockIdx.x, t = threadIdx.x;
  for (int j = t; j < n; j += 128) sv[j] = S[(size_t)b * n + j];
  __syncthreads();
  for (int i = 0; i < k; ++i) {
    float bv = -INFINITY; int bi = n;
    for (int j = t; j < n; j += 128) {
      float v = sv[j];
      if (v > bv || (v == bv && j < bi)) { bv = v; bi = j; }
    }
    rv[t] = bv; ri[t] = bi; __syncthreads();
    for (int s = 64; s > 0; s >>= 1) {
      if (t < s) {
        if (rv[t + s] > rv[t] || (rv[t + s] == rv[t] && ri[t + s] < ri[t])) {
          rv[t] = rv[t + s]; ri[t] = ri[t + s];
        }
      }
      __syncthreads();
    }
    if (t == 0) { Idx[(size_t)b * k + i] = ri[0]; Sv[(size_t)b * k + i] = rv[0]; sv[ri[0]] = -INFINITY; }
    __syncthreads();
  }
}

// ---------------- pooled gather (bf16->bf16): Hout[b,i] = Hn[b, idx] * sigmoid(sv) ----------------
__global__ void bgnn_pool_gather(const __bf16* __restrict__ Hn, const int* __restrict__ Idx,
                                 const float* __restrict__ Sv, __bf16* __restrict__ Hout,
                                 int nprev, int k) {
  int bi = blockIdx.x;             // b*k + i
  int b = bi / k;
  int f = threadIdx.x;
  int j = Idx[bi];
  float gate = sigmoidf_(Sv[bi]);
  Hout[(size_t)bi * HH + f] = f2bf(bf2f(Hn[((size_t)b * nprev + j) * HH + f]) * gate);
}

// ---------------- per-graph softmax over n scores (blockDim >= n) ----------------
__global__ void bgnn_graph_softmax(float* __restrict__ G, int n) {
  __shared__ float red[64];
  int b = blockIdx.x, t = threadIdx.x;
  float v = (t < n) ? G[(size_t)b * n + t] : -INFINITY;
  red[t] = v; __syncthreads();
  for (int s = 32; s > 0; s >>= 1) { if (t < s) red[t] = fmaxf(red[t], red[t + s]); __syncthreads(); }
  float mx = red[0]; __syncthreads();
  float e = (t < n) ? expf(v - mx) : 0.0f;
  red[t] = e; __syncthreads();
  for (int s = 32; s > 0; s >>= 1) { if (t < s) red[t] += red[t + s]; __syncthreads(); }
  float inv = 1.0f / red[0];
  if (t < n) G[(size_t)b * n + t] = e * inv;
}

// ---------------- attention pool: XGb[b,f] = bf16( sum_j Hn[b,j,f]*G[b,j] ) ----------------
__global__ void bgnn_attn_pool(const __bf16* __restrict__ Hn, const float* __restrict__ G,
                               __bf16* __restrict__ XGb, int n) {
  int b = blockIdx.x, f = threadIdx.x;
  float acc = 0.0f;
  for (int j = 0; j < n; ++j)
    acc += bf2f(Hn[((size_t)b * n + j) * HH + f]) * G[(size_t)b * n + j];
  XGb[(size_t)b * HH + f] = f2bf(acc);
}

// =====================================================================
// host-side orchestration
// =====================================================================
extern "C" void kernel_launch(void* const* d_in, const int* in_sizes, int n_in,
                              void* d_out, int out_size, void* d_ws, size_t ws_size,
                              hipStream_t stream) {
  (void)in_sizes; (void)n_in; (void)out_size; (void)ws_size;

  // ---- inputs (setup_inputs order; params leaves in JAX pytree (alphabetical) order) ----
  const float* x      = (const float*)d_in[0];
  const int*   eidx   = (const int*)  d_in[1];
  const float* eattr  = (const float*)d_in[2];
  struct Blk { const float *We,*aW1,*aW2,*ab1,*ab2,*basis,*be,*bias,*nb,*ng,*roi; };
  Blk blk[3];
  for (int l = 0; l < 3; ++l) {
    int o = 3 + l * 11;
    blk[l].We    = (const float*)d_in[o + 0];
    blk[l].aW1   = (const float*)d_in[o + 1];
    blk[l].aW2   = (const float*)d_in[o + 2];
    blk[l].ab1   = (const float*)d_in[o + 3];
    blk[l].ab2   = (const float*)d_in[o + 4];
    blk[l].basis = (const float*)d_in[o + 5];
    blk[l].be    = (const float*)d_in[o + 6];
    blk[l].bias  = (const float*)d_in[o + 7];
    blk[l].nb    = (const float*)d_in[o + 8];
    blk[l].ng    = (const float*)d_in[o + 9];
    blk[l].roi   = (const float*)d_in[o + 10];
  }
  const float* gW1    = (const float*)d_in[36];
  const float* gW2    = (const float*)d_in[37];
  const float* gb1    = (const float*)d_in[38];
  const float* gb2    = (const float*)d_in[39];
  const float* hW1    = (const float*)d_in[40];
  const float* hW2    = (const float*)d_in[41];
  const float* hW3    = (const float*)d_in[42];
  const float* hb1    = (const float*)d_in[43];
  const float* hb2    = (const float*)d_in[44];
  const float* hb3    = (const float*)d_in[45];
  const float* hbeta  = (const float*)d_in[46];
  const float* hg     = (const float*)d_in[47];
  const float* in_W   = (const float*)d_in[48];
  const float* in_b   = (const float*)d_in[49];
  const float* in_beta= (const float*)d_in[50];
  const float* in_g   = (const float*)d_in[51];

  // ---- workspace carve-up ----
  char* ws = (char*)d_ws;
  size_t off = 0;
  auto carve = [&](size_t bytes) { void* p = ws + off; off = (off + bytes + 255) & ~(size_t)255; return p; };

  __bf16* XA  = (__bf16*)carve((size_t)N0 * KP0 * 2);        // x in bf16, K padded to 288
  __bf16* FB0 = (__bf16*)carve((size_t)N0 * HH * 2);         // feature ping (bf16)
  __bf16* FB1 = (__bf16*)carve((size_t)N0 * HH * 2);         // hnorm / pong (bf16)
  float*  W1  = (float*) carve((size_t)N0 * HH * 4);         // input raw / ht (f32)
  float*  W2  = (float*) carve((size_t)N0 * HH * 4);         // pre-norm out (atomics target)
  float*  S   = (float*) carve((size_t)BG * RR * 4);
  float*  SV  = (float*) carve((size_t)BG * RR * 4);
  int*    IDX = (int*)   carve((size_t)BG * RR * 4);
  float*  CW  = (float*) carve((size_t)RR * CC * 4);
  float*  G   = (float*) carve((size_t)BG * 64 * 4);
  __bf16* XGb = (__bf16*)carve((size_t)BG * HH * 2);
  float*  Z1  = (float*) carve((size_t)BG * HH * 4);
  __bf16* Z1b = (__bf16*)carve((size_t)BG * HH * 2);
  float*  Z2  = (float*) carve((size_t)BG * 64 * 4);
  // packed bf16 weights (fragment order)
  const size_t PK128 = 4 * 8 * 512;                          // packed 128x128 -> 16384 bf16
  __bf16* Pin  = (__bf16*)carve((size_t)9 * 8 * 512 * 2);    // 288x128
  __bf16* Pbas = (__bf16*)carve((size_t)3 * CC * PK128 * 2); // 21 bases
  __bf16* PaW1 = (__bf16*)carve((size_t)3 * PK128 * 2);
  __bf16* PgW1 = (__bf16*)carve(PK128 * 2);
  __bf16* PhW1 = (__bf16*)carve(PK128 * 2);
  __bf16* PhW2 = (__bf16*)carve((size_t)4 * 4 * 512 * 2);    // 128x64

  const int KS[3] = {214, 128, 51};

  // ---- weight packing + input conversion (cheap, once per call) ----
  bgnn_pack_b<<<8 * 9, 32, 0, stream>>>(in_W, Pin, RR, KP0, HH);
  for (int l = 0; l < 3; ++l) {
    for (int c = 0; c < CC; ++c)
      bgnn_pack_b<<<8 * 4, 32, 0, stream>>>(blk[l].basis + (size_t)c * HH * HH,
                                            Pbas + ((size_t)l * CC + c) * PK128, HH, HH, HH);
    bgnn_pack_b<<<8 * 4, 32, 0, stream>>>(blk[l].aW1, PaW1 + (size_t)l * PK128, HH, HH, HH);
  }
  bgnn_pack_b<<<8 * 4, 32, 0, stream>>>(gW1, PgW1, HH, HH, HH);
  bgnn_pack_b<<<8 * 4, 32, 0, stream>>>(hW1, PhW1, HH, HH, HH);
  bgnn_pack_b<<<4 * 4, 32, 0, stream>>>(hW2, PhW2, HH, HH, 64);
  bgnn_cvt_x<<<N0, KP0, 0, stream>>>(x, XA);

  // ---- input projection: W1 = x @ in_W + in_b ; FB0 = bf16(elu(ln(W1))) ----
  bgnn_wmma_gemm<0><<<N0 / 16, 32, 0, stream>>>(XA, Pin, in_b, W1, KP0, HH);
  bgnn_actln<0><<<N0, HH, 0, stream>>>(W1, in_g, in_beta, FB0);

  // ---- GNN blocks ----
  int nprev = RR;
  for (int l = 0; l < 3; ++l) {
    int M = BG * nprev;                                   // always divisible by 16
    bgnn_row_softmax7<<<(RR + 127) / 128, 128, 0, stream>>>(blk[l].roi, CW);

    bgnn_wmma_transform<<<M / 16, 32, 0, stream>>>(FB0, Pbas + (size_t)l * CC * PK128, CW,
                                                   blk[l].We, blk[l].be, blk[l].bias,
                                                   W1, W2, (l == 0) ? 1 : 0);
    if (l == 0)
      bgnn_edge_agg<<<EE, HH, 0, stream>>>(eidx, eattr, W1, blk[l].We, blk[l].be, W2);

    bgnn_actln<1><<<M, HH, 0, stream>>>(W2, blk[l].ng, blk[l].nb, FB1);

    // fused pooling attention score: S = tanh(FB1 @ aW1 + ab1) . aW2 + ab2
    bgnn_wmma_score<<<M / 16, 32, 0, stream>>>(FB1, PaW1 + (size_t)l * PK128,
                                               blk[l].ab1, blk[l].aW2, blk[l].ab2, S);

    int k = KS[l];
    bgnn_topk<<<BG, 128, 0, stream>>>(S, nprev, k, IDX, SV);
    bgnn_pool_gather<<<BG * k, HH, 0, stream>>>(FB1, IDX, SV, FB0, nprev, k);
    nprev = k;
  }

  // ---- head: global attention pool + MLP ----
  int M3 = BG * nprev;  // 256*51 = 13056
  bgnn_wmma_score<<<M3 / 16, 32, 0, stream>>>(FB0, PgW1, gb1, gW2, gb2, G);
  bgnn_graph_softmax<<<BG, 64, 0, stream>>>(G, nprev);
  bgnn_attn_pool<<<BG, HH, 0, stream>>>(FB0, G, XGb, nprev);

  bgnn_wmma_gemm<0><<<BG / 16, 32, 0, stream>>>(XGb, PhW1, hb1, Z1, HH, HH);
  bgnn_actln<0><<<BG, HH, 0, stream>>>(Z1, hg, hbeta, Z1b);
  bgnn_wmma_gemm<2><<<BG / 16, 32, 0, stream>>>(Z1b, PhW2, hb2, Z2, HH, 64);
  bgnn_rowdot<<<BG, 64, 0, stream>>>(Z2, hW3, hb3, (float*)d_out, 64);
}
